// Sam_Backbone_5935644803373
// MI455X (gfx1250) — matrix operand order repaired and checked by reading, not compiled, not run on hardware
//
#include <hip/hip_runtime.h>
#include <math.h>

// ---------------------------------------------------------------------------
// SAM ViT-B backbone on gfx1250 (CDNA5).
//  - All GEMMs on v_wmma_f32_16x16x32_bf16 (f32 accumulate).
//  - Interior GEMM tiles streamed by the Tensor Data Mover with DOUBLE
//    BUFFERING: issue tile k+1, s_wait_tensorcnt 2 (tile k landed, k+1 in
//    flight during the WMMAs). Boundary tiles via clamped loads.
//  - GEMM: 128x64 macro tile, 8 waves, each wave 32x32 (4 WMMAs / K-step).
// ---------------------------------------------------------------------------

typedef float  v8f   __attribute__((ext_vector_type(8)));
typedef __bf16 v16bf __attribute__((ext_vector_type(16)));
typedef unsigned short v16u __attribute__((ext_vector_type(16)));
typedef unsigned int u32x4 __attribute__((ext_vector_type(4)));
typedef int i32x4 __attribute__((ext_vector_type(4)));
typedef int i32x8 __attribute__((ext_vector_type(8)));

#define TDM_AVAILABLE (__has_builtin(__builtin_amdgcn_tensor_load_to_lds))

__device__ __forceinline__ unsigned short f2bf(float f) {
    unsigned int u = __float_as_uint(f);
    unsigned int r = u + 0x7FFFu + ((u >> 16) & 1u);   // round-to-nearest-even
    return (unsigned short)(r >> 16);
}

// Issue one 2-D TDM load: tile (tx elems) x (ty rows) of fp32, row stride
// strideElems, global -> LDS byte offset ldsAddr. D# per CDNA5 ISA ch.8.
__device__ __forceinline__ void tdm_load_2d(unsigned ldsAddr, const void* gptr,
                                            unsigned tx, unsigned ty,
                                            unsigned long strideElems)
{
#if TDM_AVAILABLE
    unsigned long ga = (unsigned long)gptr;
    u32x4 g0;
    g0[0] = 1u;                                            // count=1, user desc
    g0[1] = ldsAddr;                                       // lds_addr (bytes)
    g0[2] = (unsigned)ga;                                  // global_addr[31:0]
    g0[3] = (unsigned)((ga >> 32) & 0x1FFFFFFu) | (2u << 30); // addr[56:32], type=2
    i32x8 g1;
    g1[0] = 0x00020000;                                    // data_size=2 (4B)
    g1[1] = (int)((tx & 0xFFFFu) << 16);                   // tensor_dim0[15:0]
    g1[2] = (int)(((ty & 0xFFFFu) << 16) | (tx >> 16));    // td1[15:0] | td0[31:16]
    g1[3] = (int)(((tx & 0xFFFFu) << 16) | (ty >> 16));    // tile_dim0 | td1[31:16]
    g1[4] = (int)(ty & 0xFFFFu);                           // tile_dim1 (tile_dim2=0)
    g1[5] = (int)(strideElems & 0xFFFFFFFFu);              // tensor_dim0_stride lo
    g1[6] = (int)((strideElems >> 32) & 0xFFFFu);          // stride hi16
    g1[7] = 0;
    i32x4 z4 = {0, 0, 0, 0};
#if defined(__clang_major__) && (__clang_major__ >= 23)
    i32x8 z8 = {0, 0, 0, 0, 0, 0, 0, 0};
    __builtin_amdgcn_tensor_load_to_lds(g0, g1, z4, z4, z8, 0);
#else
    __builtin_amdgcn_tensor_load_to_lds(g0, g1, z4, z4, 0);
#endif
#else
    (void)ldsAddr; (void)gptr; (void)tx; (void)ty; (void)strideElems;
#endif
}

// Build bf16 fragments from an fp32 LDS tile pair and do the 4 WMMAs.
__device__ __forceinline__ void tile_mma(const float (*Afb)[32], const float (*Bfb)[32],
                                         int transB, int wm, int wn, int half, int lrow,
                                         v8f& acc00, v8f& acc01, v8f& acc10, v8f& acc11)
{
    const float (*Bt)[64] = (const float(*)[64])&Bfb[0][0];
    v16u au0, au1, bu0, bu1;
    const int ar0 = wm * 32 + lrow;
    const int ar1 = ar0 + 16;
    #pragma unroll
    for (int e = 0; e < 16; ++e) {
        int kk = 16 * half + e;                        // A: K = 16*half + e
        au0[e] = f2bf(Afb[ar0][kk]);
        au1[e] = f2bf(Afb[ar1][kk]);
    }
    const int bc0 = wn * 32 + lrow;
    const int bc1 = bc0 + 16;
    if (!transB) {
        #pragma unroll
        for (int e = 0; e < 16; ++e) {
            int kk = 16 * half + e;                    // B: K = 16*half + e
            bu0[e] = f2bf(Bfb[bc0][kk]);
            bu1[e] = f2bf(Bfb[bc1][kk]);
        }
    } else {
        #pragma unroll
        for (int e = 0; e < 16; ++e) {
            int kk = 16 * half + e;
            bu0[e] = f2bf(Bt[kk][bc0]);
            bu1[e] = f2bf(Bt[kk][bc1]);
        }
    }
    v16bf a0 = __builtin_bit_cast(v16bf, au0);
    v16bf a1 = __builtin_bit_cast(v16bf, au1);
    v16bf b0 = __builtin_bit_cast(v16bf, bu0);
    v16bf b1 = __builtin_bit_cast(v16bf, bu1);
    acc00 = __builtin_amdgcn_wmma_f32_16x16x32_bf16(false, a0, false, b0, (short)0, acc00, false, false);
    acc01 = __builtin_amdgcn_wmma_f32_16x16x32_bf16(false, a0, false, b1, (short)0, acc01, false, false);
    acc10 = __builtin_amdgcn_wmma_f32_16x16x32_bf16(false, a1, false, b0, (short)0, acc10, false, false);
    acc11 = __builtin_amdgcn_wmma_f32_16x16x32_bf16(false, a1, false, b1, (short)0, acc11, false, false);
}

// ---------------------------------------------------------------------------
// Batched GEMM:  C[z] = act( alpha * A[z] (MxK) @ op(B[z]) + bias + resid )
//   A batch offset = (z/innerB)*aOuter + (z%innerB)*aInner   (same for B)
//   transB==0: B is (N,K) row-major (out = A @ B^T);  transB==1: B is (K,N).
// ---------------------------------------------------------------------------
__global__ __launch_bounds__(256) void gemm_wmma_kernel(
    const float* __restrict__ A, long aOuter, long aInner, int lda,
    const float* __restrict__ B, long bOuter, long bInner, int ldb, int transB,
    const float* __restrict__ bias,
    const float* __restrict__ resid, long rStride,
    float* __restrict__ C, long cStride, int ldc,
    int M, int N, int K, int innerB, float alpha, int act)
{
    __shared__ float Af[2][128][32];     // A tiles [buf][m][k], 32KB
    __shared__ float Bf[2][64][32];      // B tiles: [n][k] (notrans) / [k][n] (trans), 16KB

    const int z = blockIdx.z;
    const float* Ab = A + (long)(z / innerB) * aOuter + (long)(z % innerB) * aInner;
    const float* Bb = B + (long)(z / innerB) * bOuter + (long)(z % innerB) * bInner;
    float* Cb = C + (long)z * cStride;
    const float* Rb = resid ? (resid + (long)z * rStride) : nullptr;

    const int m0 = blockIdx.y * 128;
    const int n0 = blockIdx.x * 64;
    const int tid  = threadIdx.x;
    const int wave = tid >> 5;
    const int lane = tid & 31;
    const int half = lane >> 4;
    const int lrow = lane & 15;
    const int wm = wave >> 1;   // 0..3 -> 32-row group
    const int wn = wave & 1;    // 0..1 -> 32-col group

    const bool fullMN = (m0 + 128 <= M) && (n0 + 64 <= N);

    v8f zero = {0,0,0,0,0,0,0,0};
    v8f acc00 = zero, acc01 = zero, acc10 = zero, acc11 = zero;

#if TDM_AVAILABLE
    if (fullMN && ((K & 31) == 0)) {
        // ---- pipelined TDM path: double-buffered LDS, DMA overlaps WMMA ----
        const int nk = K >> 5;
        if (wave == 0) {
            tdm_load_2d((unsigned)(unsigned long)&Af[0][0][0],
                        Ab + (long)m0 * lda, 32u, 128u, (unsigned long)lda);
            if (!transB)
                tdm_load_2d((unsigned)(unsigned long)&Bf[0][0][0],
                            Bb + (long)n0 * ldb, 32u, 64u, (unsigned long)ldb);
            else
                tdm_load_2d((unsigned)(unsigned long)&Bf[0][0][0],
                            Bb + n0, 64u, 32u, (unsigned long)ldb);
        }
        for (int kt = 0; kt < nk; ++kt) {
            const int buf = kt & 1;
            if (wave == 0) {
                if (kt + 1 < nk) {
                    const int k1 = (kt + 1) << 5;
                    const int nb = buf ^ 1;
                    tdm_load_2d((unsigned)(unsigned long)&Af[nb][0][0],
                                Ab + (long)m0 * lda + k1, 32u, 128u, (unsigned long)lda);
                    if (!transB)
                        tdm_load_2d((unsigned)(unsigned long)&Bf[nb][0][0],
                                    Bb + (long)n0 * ldb + k1, 32u, 64u, (unsigned long)ldb);
                    else
                        tdm_load_2d((unsigned)(unsigned long)&Bf[nb][0][0],
                                    Bb + (long)k1 * ldb + n0, 64u, 32u, (unsigned long)ldb);
                    __builtin_amdgcn_s_wait_tensorcnt(2);   // tile kt complete (in-order)
                } else {
                    __builtin_amdgcn_s_wait_tensorcnt(0);
                }
            }
            __syncthreads();
            tile_mma(Af[buf], Bf[buf], transB, wm, wn, half, lrow, acc00, acc01, acc10, acc11);
            __syncthreads();
        }
    } else
#endif
    {
        // ---- generic path (boundary blocks / K not multiple of 32) ----
        float (*Bt0)[64] = (float(*)[64])&Bf[0][0][0];
        for (int k0 = 0; k0 < K; k0 += 32) {
            const bool fast = fullMN && (k0 + 32 <= K);
            if (fast) {
                #pragma unroll
                for (int j = 0; j < 4; ++j) {           // A: 128x32 = 4 float4/thread
                    int idx = tid + j * 256;
                    int r = idx >> 3;
                    int cg = (idx & 7) * 4;
                    float4 v = *(const float4*)(Ab + (long)(m0 + r) * lda + (k0 + cg));
                    *(float4*)&Af[0][r][cg] = v;
                }
                if (!transB) {
                    #pragma unroll
                    for (int j = 0; j < 2; ++j) {       // B: 64x32 = 2 float4/thread
                        int idx = tid + j * 256;
                        int r = idx >> 3;
                        int cg = (idx & 7) * 4;
                        float4 v = *(const float4*)(Bb + (long)(n0 + r) * ldb + (k0 + cg));
                        *(float4*)&Bf[0][r][cg] = v;
                    }
                } else {
                    #pragma unroll
                    for (int j = 0; j < 2; ++j) {       // B (KxN): contiguous along n
                        int idx = tid + j * 256;
                        int k = idx >> 4;
                        int ng = (idx & 15) * 4;
                        float4 v = *(const float4*)(Bb + (long)(k0 + k) * ldb + (n0 + ng));
                        *(float4*)&Bt0[k][ng] = v;
                    }
                }
            } else {
                // boundary: clamp address, select zero (no exec branches)
                #pragma unroll
                for (int e = 0; e < 16; ++e) {
                    int idx = tid + e * 256;
                    int r = idx >> 5, c = idx & 31;
                    int gm = m0 + r, gk = k0 + c;
                    int cm = gm < M ? gm : M - 1;
                    int ck = gk < K ? gk : K - 1;
                    float v = Ab[(long)cm * lda + ck];
                    Af[0][r][c] = (gm < M && gk < K) ? v : 0.f;
                }
                #pragma unroll
                for (int e = 0; e < 8; ++e) {
                    int idx = tid + e * 256;
                    int r = idx >> 5, c = idx & 31;   // r = n, c = k
                    int gn = n0 + r, gk = k0 + c;
                    int cn = gn < N ? gn : N - 1;
                    int ck = gk < K ? gk : K - 1;
                    float v = transB ? Bb[(long)ck * ldb + cn] : Bb[(long)cn * ldb + ck];
                    v = (gn < N && gk < K) ? v : 0.f;
                    if (transB) Bt0[c][r] = v; else Bf[0][r][c] = v;
                }
            }
            __syncthreads();
            tile_mma(Af[0], Bf[0], transB, wm, wn, half, lrow, acc00, acc01, acc10, acc11);
            __syncthreads();
        }
    }

    // Epilogue. C/D layout: lane l: n = l%16, VGPR r: m = r + 8*(l/16).
    #pragma unroll
    for (int ai = 0; ai < 2; ++ai) {
        #pragma unroll
        for (int bj = 0; bj < 2; ++bj) {
            v8f acc = ai ? (bj ? acc11 : acc10) : (bj ? acc01 : acc00);
            #pragma unroll
            for (int r = 0; r < 8; ++r) {
                int m = m0 + wm * 32 + ai * 16 + r + 8 * half;
                int n = n0 + wn * 32 + bj * 16 + lrow;
                if (m >= M || n >= N) continue;
                float v = alpha * acc[r];
                if (bias) v += bias[n];
                if (Rb)   v += Rb[(long)m * ldc + n];
                if (act == 1) v = 0.5f * v * (1.0f + erff(v * 0.70710678118f));  // exact GELU
                Cb[(long)m * ldc + n] = v;
            }
        }
    }
}

// ---------------------------------------------------------------------------
// LayerNorm over last dim (also LN2d per-token); transOut writes NCHW.
// ---------------------------------------------------------------------------
__global__ __launch_bounds__(256) void ln_kernel(
    const float* __restrict__ x, const float* __restrict__ w, const float* __restrict__ b,
    float* __restrict__ y, int C, int rows, int transOut)
{
    __shared__ float red[256];
    const int row = blockIdx.x;
    const float* xr = x + (long)row * C;
    const int tid = threadIdx.x;
    float s = 0.f;
    for (int c = tid; c < C; c += 256) s += xr[c];
    red[tid] = s; __syncthreads();
    for (int o = 128; o > 0; o >>= 1) { if (tid < o) red[tid] += red[tid + o]; __syncthreads(); }
    float mean = red[0] / (float)C; __syncthreads();
    float v = 0.f;
    for (int c = tid; c < C; c += 256) { float d = xr[c] - mean; v += d * d; }
    red[tid] = v; __syncthreads();
    for (int o = 128; o > 0; o >>= 1) { if (tid < o) red[tid] += red[tid + o]; __syncthreads(); }
    float rstd = rsqrtf(red[0] / (float)C + 1e-6f);
    for (int c = tid; c < C; c += 256) {
        float o = (xr[c] - mean) * rstd * w[c] + b[c];
        if (transOut) y[(long)c * rows + row] = o;
        else          y[(long)row * C + c] = o;
    }
}

// pad 32x32 -> 42x42 and split into 9 windows of 14x14 (zeros in padding)
__global__ void win_partition_kernel(const float* __restrict__ t, float* __restrict__ tw)
{
    int blk = blockIdx.x;                 // win*196 + i
    int win = blk / 196, i = blk % 196;
    int h = (win / 3) * 14 + i / 14;
    int w = (win % 3) * 14 + i % 14;
    bool in = (h < 32) && (w < 32);
    const float* src = t + (long)(h * 32 + w) * 768;
    float* dst = tw + (long)blk * 768;
    for (int c = threadIdx.x; c < 768; c += blockDim.x)
        dst[c] = in ? src[c] : 0.f;
}

// merge heads + window-unpartition -> (1024, 768)
__global__ void merge_kernel(const float* __restrict__ ao, float* __restrict__ out, int windowed)
{
    int token = blockIdx.x; int h = token >> 5, w = token & 31;
    for (int ch = threadIdx.x; ch < 768; ch += blockDim.x) {
        int head = ch >> 6, c = ch & 63;
        float v;
        if (windowed) {
            int win = (h / 14) * 3 + (w / 14);
            int i = (h % 14) * 14 + (w % 14);
            v = ao[(((long)(win * 12 + head)) * 196 + i) * 64 + c];
        } else {
            v = ao[((long)head * 1024 + token) * 64 + c];
        }
        out[(long)token * 768 + ch] = v;
    }
}

// decomposed rel-pos bias: bias[bh,i,k] = q[bh,i,:] . rel[pos-k+S-1,:]
__global__ void rel_bias_kernel(const float* __restrict__ qkv, long winStride, int heads,
                                const float* __restrict__ rel, float* __restrict__ bias,
                                int H, int W, int S, int mode)
{
    int i  = blockIdx.x;                  // token within window
    int bh = blockIdx.y;                  // win*heads + head
    int win = bh / heads, head = bh % heads;
    const float* q = qkv + (long)win * winStride + (long)i * 2304 + head * 64;
    int pos = mode ? (i % W) : (i / W);
    float* out = bias + ((long)bh * (H * W) + i) * S;
    for (int k = threadIdx.x; k < S; k += blockDim.x) {
        const float* R = rel + (long)(pos - k + S - 1) * 64;
        float d = 0.f;
        #pragma unroll
        for (int c = 0; c < 64; ++c) d += q[c] * R[c];
        out[k] = d;
    }
}

// softmax over keys with fused decomposed bias add (in place)
__global__ __launch_bounds__(256) void softmax_bias_kernel(
    float* __restrict__ scores, const float* __restrict__ biasH,
    const float* __restrict__ biasW, int H, int W, int N)
{
    __shared__ float red[256];
    long row = blockIdx.x;                // bh*HW + i
    int HW = H * W;
    long bh = row / HW;
    int i = (int)(row % HW);
    float* s = scores + row * (long)N;
    const float* bH = biasH + (bh * HW + i) * (long)H;
    const float* bW = biasW + (bh * HW + i) * (long)W;
    int tid = threadIdx.x;
    float mx = -3.4e38f;
    for (int j = tid; j < N; j += 256) {
        float v = s[j] + bH[j / W] + bW[j % W];
        s[j] = v;
        mx = fmaxf(mx, v);
    }
    red[tid] = mx; __syncthreads();
    for (int o = 128; o > 0; o >>= 1) { if (tid < o) red[tid] = fmaxf(red[tid], red[tid + o]); __syncthreads(); }
    mx = red[0]; __syncthreads();
    float sum = 0.f;
    for (int j = tid; j < N; j += 256) { float e = expf(s[j] - mx); s[j] = e; sum += e; }
    red[tid] = sum; __syncthreads();
    for (int o = 128; o > 0; o >>= 1) { if (tid < o) red[tid] += red[tid + o]; __syncthreads(); }
    float inv = 1.f / red[0];
    for (int j = tid; j < N; j += 256) s[j] *= inv;
}

// im2col for 16x16 stride-16 patch embed: (1024, 768), k = c*256 + i*16 + j
__global__ void im2col_patch_kernel(const float* __restrict__ x, float* __restrict__ A)
{
    int token = blockIdx.x; int th = token >> 5, tw = token & 31;
    for (int k = threadIdx.x; k < 768; k += blockDim.x) {
        int c = k >> 8, rem = k & 255, i = rem >> 4, j = rem & 15;
        A[(long)token * 768 + k] = x[(long)c * 512 * 512 + (long)(th * 16 + i) * 512 + (tw * 16 + j)];
    }
}

// im2col for 3x3 SAME conv on 32x32x256: (1024, 2304), k = c*9 + ky*3 + kx
__global__ void im2col3_kernel(const float* __restrict__ y, float* __restrict__ col)
{
    int token = blockIdx.x; int h = token >> 5, w = token & 31;
    for (int k = threadIdx.x; k < 2304; k += blockDim.x) {
        int c = k / 9, r = k % 9, ky = r / 3, kx = r % 3;
        int hh = h + ky - 1, ww = w + kx - 1;
        float v = (hh >= 0 && hh < 32 && ww >= 0 && ww < 32) ? y[(long)(hh * 32 + ww) * 256 + c] : 0.f;
        col[(long)token * 2304 + k] = v;
    }
}

// ---------------------------------------------------------------------------
static void launch_gemm(hipStream_t stream,
    const float* A, long aO, long aI, int lda,
    const float* B, long bO, long bI, int ldb, int transB,
    const float* bias, const float* resid, long rStride,
    float* C, long cStride, int ldc,
    int M, int N, int K, int batches, int innerB, float alpha, int act)
{
    dim3 grid((N + 63) / 64, (M + 127) / 128, batches);
    gemm_wmma_kernel<<<grid, 256, 0, stream>>>(A, aO, aI, lda, B, bO, bI, ldb, transB,
                                               bias, resid, rStride, C, cStride, ldc,
                                               M, N, K, innerB, alpha, act);
}

extern "C" void kernel_launch(void* const* d_in, const int* in_sizes, int n_in,
                              void* d_out, int out_size, void* d_ws, size_t ws_size,
                              hipStream_t stream) {
    (void)in_sizes; (void)n_in; (void)out_size; (void)ws_size;
    const float* x         = (const float*)d_in[0];
    const float* patch_w   = (const float*)d_in[1];
    const float* patch_b   = (const float*)d_in[2];
    const float* pos_embed = (const float*)d_in[3];
    const float* qkv_w     = (const float*)d_in[4];
    const float* qkv_b     = (const float*)d_in[5];
    const float* proj_w    = (const float*)d_in[6];
    const float* proj_b    = (const float*)d_in[7];
    const float* norm1_w   = (const float*)d_in[8];
    const float* norm1_b   = (const float*)d_in[9];
    const float* norm2_w   = (const float*)d_in[10];
    const float* norm2_b   = (const float*)d_in[11];
    const float* mlp_w1    = (const float*)d_in[12];
    const float* mlp_b1    = (const float*)d_in[13];
    const float* mlp_w2    = (const float*)d_in[14];
    const float* mlp_b2    = (const float*)d_in[15];
    const float* rel_h_win = (const float*)d_in[16];
    const float* rel_w_win = (const float*)d_in[17];
    const float* rel_h_glob= (const float*)d_in[18];
    const float* rel_w_glob= (const float*)d_in[19];
    const float* neck_w1   = (const float*)d_in[20];
    const float* neck_ln1_w= (const float*)d_in[21];
    const float* neck_ln1_b= (const float*)d_in[22];
    const float* neck_w2   = (const float*)d_in[23];
    const float* neck_ln2_w= (const float*)d_in[24];
    const float* neck_ln2_b= (const float*)d_in[25];

    // workspace bump allocator
    char* p = (char*)d_ws;
    auto alloc = [&](long n) -> float* {
        float* r = (float*)p;
        p += ((n * (long)sizeof(float) + 255) / 256) * 256;
        return r;
    };
    float* h      = alloc(1024L * 768);
    float* tln    = alloc(1024L * 768);
    float* twin   = alloc(1764L * 768);     // windowed tokens (also reused as y2)
    float* qkv    = alloc(1764L * 2304);
    float* bH     = alloc(12L * 1024 * 32); // max over windowed/global
    float* bW     = alloc(12L * 1024 * 32);
    float* scores = alloc(12L * 1024 * 1024);
    float* aout   = alloc(108L * 196 * 64);
    float* merged = alloc(1024L * 768);     // reused as neck y1
    float* mlpb   = alloc(1024L * 3072);    // reused as neck im2col

    // ---- patch embed (im2col + GEMM, bias + pos_embed fused) ----
    im2col_patch_kernel<<<1024, 256, 0, stream>>>(x, twin);
    launch_gemm(stream, twin, 0, 0, 768, patch_w, 0, 0, 768, 0,
                patch_b, pos_embed, 0, h, 0, 768, 1024, 768, 768, 1, 1, 1.f, 0);

    for (int i = 0; i < 12; ++i) {
        const bool glob = (i == 2 || i == 5 || i == 8 || i == 11);
        const int S    = glob ? 32 : 14;
        const int Ntok = glob ? 1024 : 196;
        const int nwin = glob ? 1 : 9;
        const int BHc  = nwin * 12;
        const float* relh = glob ? rel_h_glob + (long)i * 63 * 64 : rel_h_win + (long)i * 27 * 64;
        const float* relw = glob ? rel_w_glob + (long)i * 63 * 64 : rel_w_win + (long)i * 27 * 64;

        // LN1
        ln_kernel<<<1024, 256, 0, stream>>>(h, norm1_w + i * 768, norm1_b + i * 768, tln, 768, 1024, 0);
        const float* attn_in = tln;
        if (!glob) { win_partition_kernel<<<1764, 256, 0, stream>>>(tln, twin); attn_in = twin; }
        const int Mrows = nwin * Ntok;

        // QKV projection
        launch_gemm(stream, attn_in, 0, 0, 768, qkv_w + (long)i * 2304 * 768, 0, 0, 768, 0,
                    qkv_b + (long)i * 2304, nullptr, 0, qkv, 0, 2304, Mrows, 2304, 768, 1, 1, 1.f, 0);

        // decomposed rel-pos biases
        rel_bias_kernel<<<dim3(Ntok, BHc), 32, 0, stream>>>(qkv, (long)Ntok * 2304, 12, relh, bH, S, S, S, 0);
        rel_bias_kernel<<<dim3(Ntok, BHc), 32, 0, stream>>>(qkv, (long)Ntok * 2304, 12, relw, bW, S, S, S, 1);

        // scores = 0.125 * q @ k^T   (batched over win x head)
        launch_gemm(stream, qkv, (long)Ntok * 2304, 64, 2304,
                    qkv + 768, (long)Ntok * 2304, 64, 2304, 0,
                    nullptr, nullptr, 0,
                    scores, (long)Ntok * Ntok, Ntok,
                    Ntok, Ntok, 64, BHc, 12, 0.125f, 0);

        // softmax with fused bias add
        softmax_bias_kernel<<<BHc * Ntok, 256, 0, stream>>>(scores, bH, bW, S, S, Ntok);

        // out = attn @ v   (B is K x N -> transB)
        launch_gemm(stream, scores, 12L * Ntok * Ntok, (long)Ntok * Ntok, Ntok,
                    qkv + 1536, (long)Ntok * 2304, 64, 2304, 1,
                    nullptr, nullptr, 0,
                    aout, (long)Ntok * 64, 64,
                    Ntok, 64, Ntok, BHc, 12, 1.f, 0);

        // merge heads (+ window unpartition)
        merge_kernel<<<1024, 256, 0, stream>>>(aout, merged, glob ? 0 : 1);

        // proj + residual
        launch_gemm(stream, merged, 0, 0, 768, proj_w + (long)i * 768 * 768, 0, 0, 768, 0,
                    proj_b + (long)i * 768, h, 0, h, 0, 768, 1024, 768, 768, 1, 1, 1.f, 0);

        // MLP: LN2 -> fc1 (GELU) -> fc2 + residual
        ln_kernel<<<1024, 256, 0, stream>>>(h, norm2_w + i * 768, norm2_b + i * 768, tln, 768, 1024, 0);
        launch_gemm(stream, tln, 0, 0, 768, mlp_w1 + (long)i * 3072 * 768, 0, 0, 768, 0,
                    mlp_b1 + (long)i * 3072, nullptr, 0, mlpb, 0, 3072, 1024, 3072, 768, 1, 1, 1.f, 1);
        launch_gemm(stream, mlpb, 0, 0, 3072, mlp_w2 + (long)i * 768 * 3072, 0, 0, 3072, 0,
                    mlp_b2 + (long)i * 768, h, 0, h, 0, 768, 1024, 768, 3072, 1, 1, 1.f, 0);
    }

    // ---- neck ----
    float* y1  = merged;   // (1024, 256)
    float* col = mlpb;     // (1024, 2304)
    float* y2  = twin;     // (1024, 256)
    launch_gemm(stream, h, 0, 0, 768, neck_w1, 0, 0, 768, 0,
                nullptr, nullptr, 0, y1, 0, 256, 1024, 256, 768, 1, 1, 1.f, 0);
    ln_kernel<<<1024, 256, 0, stream>>>(y1, neck_ln1_w, neck_ln1_b, tln, 256, 1024, 0);
    im2col3_kernel<<<1024, 256, 0, stream>>>(tln, col);
    launch_gemm(stream, col, 0, 0, 2304, neck_w2, 0, 0, 2304, 0,
                nullptr, nullptr, 0, y2, 0, 256, 1024, 256, 2304, 1, 1, 1.f, 0);
    // final LN2d written transposed -> NCHW (1,256,32,32)
    ln_kernel<<<1024, 256, 0, stream>>>(y2, neck_ln2_w, neck_ln2_b, (float*)d_out, 256, 1024, 1);
}
